// PointNetPropagation_42236708389456
// MI455X (gfx1250) — matrix operand order repaired and checked by reading, not compiled, not run on hardware
//
#include <hip/hip_runtime.h>
#include <hip/hip_bf16.h>

typedef __attribute__((ext_vector_type(16))) _Float16 v16h;
typedef __attribute__((ext_vector_type(8)))  _Float16 v8h;
typedef __attribute__((ext_vector_type(8)))  float    v8f;

#define BATCH 16
#define NPTS  4096
#define SPTS  1024
#define DFEAT 256
#define MROWS (BATCH * NPTS)   // 65536

// ---------------------------------------------------------------------------
// Fold BN (inference) + conv bias into per-channel scale/shift:
//   y = (conv + b - rm) * g/sqrt(rv+eps) + be  ==  conv*scale + shift
// ---------------------------------------------------------------------------
__global__ void bn_consts_kernel(const float* __restrict__ b, const float* __restrict__ g,
                                 const float* __restrict__ be, const float* __restrict__ rm,
                                 const float* __restrict__ rv, float* __restrict__ scale,
                                 float* __restrict__ shift) {
    int i = threadIdx.x;  // 256 channels
    float s = g[i] * rsqrtf(rv[i] + 1e-5f);
    scale[i] = s;
    shift[i] = (b[i] - rm[i]) * s + be[i];
}

__global__ void f32_to_f16_kernel(const float* __restrict__ in, _Float16* __restrict__ out, int n) {
    int i = blockIdx.x * blockDim.x + threadIdx.x;
    if (i < n) out[i] = (_Float16)in[i];
}

// ---------------------------------------------------------------------------
// 3-NN inverse-distance interpolation + concat.  One wave (32 lanes) per query
// point, 8 points per 256-thread block.  xyz2 of this batch staged in LDS.
// Output: X1[m][0:256]=features1 row (f16), X1[m][256:512]=interpolated (f16).
// ---------------------------------------------------------------------------
__global__ __launch_bounds__(256)
void interp_concat_kernel(const float* __restrict__ xyz1, const float* __restrict__ xyz2,
                          const float* __restrict__ f1, const float* __restrict__ f2,
                          _Float16* __restrict__ X1) {
    __shared__ float sxyz[SPTS * 3];             // 12 KB
    const int tid  = threadIdx.x;
    const int lane = tid & 31;
    const int wave = tid >> 5;
    const int b     = blockIdx.x >> 9;           // 512 blocks per batch
    const int group = blockIdx.x & 511;
    const int n     = group * 8 + wave;

    for (int i = tid; i < SPTS * 3; i += 256)
        sxyz[i] = xyz2[(size_t)b * (SPTS * 3) + i];
    __syncthreads();

    const size_t qbase = ((size_t)b * NPTS + n) * 3;
    const float qx = xyz1[qbase + 0], qy = xyz1[qbase + 1], qz = xyz1[qbase + 2];

    float bd0 = 3.4e38f, bd1 = 3.4e38f, bd2 = 3.4e38f;
    int   bi0 = 0, bi1 = 0, bi2 = 0;
    auto ins = [&](float d, int i) {
        if (d < bd2) {
            if (d < bd1) {
                if (d < bd0) { bd2 = bd1; bi2 = bi1; bd1 = bd0; bi1 = bi0; bd0 = d; bi0 = i; }
                else         { bd2 = bd1; bi2 = bi1; bd1 = d;  bi1 = i; }
            } else           { bd2 = d;  bi2 = i; }
        }
    };

    for (int s = lane; s < SPTS; s += 32) {
        float dx = sxyz[3 * s + 0] - qx;
        float dy = sxyz[3 * s + 1] - qy;
        float dz = sxyz[3 * s + 2] - qz;
        ins(fmaf(dx, dx, fmaf(dy, dy, dz * dz)), s);
    }
    // butterfly merge of per-lane top-3 lists: all lanes converge to global top-3
    for (int m = 16; m >= 1; m >>= 1) {
        float od0 = __shfl_xor(bd0, m), od1 = __shfl_xor(bd1, m), od2 = __shfl_xor(bd2, m);
        int   oi0 = __shfl_xor(bi0, m), oi1 = __shfl_xor(bi1, m), oi2 = __shfl_xor(bi2, m);
        ins(od0, oi0); ins(od1, oi1); ins(od2, oi2);
    }

    float r0 = 1.0f / (bd0 + 1e-8f);
    float r1 = 1.0f / (bd1 + 1e-8f);
    float r2 = 1.0f / (bd2 + 1e-8f);
    float rs = 1.0f / (r0 + r1 + r2);
    const float w0 = r0 * rs, w1 = r1 * rs, w2 = r2 * rs;

    const size_t m  = (size_t)b * NPTS + n;
    const size_t fb = ((size_t)b * NPTS + n) * DFEAT;
    const size_t g0 = ((size_t)b * SPTS + bi0) * DFEAT;
    const size_t g1 = ((size_t)b * SPTS + bi1) * DFEAT;
    const size_t g2 = ((size_t)b * SPTS + bi2) * DFEAT;
    for (int c = lane; c < DFEAT; c += 32) {
        X1[m * 512 + c] = (_Float16)f1[fb + c];
        float v = w0 * f2[g0 + c] + w1 * f2[g1 + c] + w2 * f2[g2 + c];
        X1[m * 512 + DFEAT + c] = (_Float16)v;
    }
}

// ---------------------------------------------------------------------------
// WMMA GEMM:  out[m][o] = act( (sum_k X[m][k]*W[o][k]) * scale[o] + shift[o] )
// Block tile 128x128, 8 waves in 4x2 grid, wave tile 32x64 (2x4 fragments of
// 16x16), K-step 32 via v_wmma_f32_16x16x32_f16.  Double-buffered LDS tiles
// with register staging (one barrier per k-step); +8-half padding makes the
// ds_load_b128 fragment reads bank-conflict free.
// ---------------------------------------------------------------------------
__device__ inline v16h make_frag(const _Float16* lo, const _Float16* hi) {
    v8h a = *(const v8h*)lo;
    v8h b = *(const v8h*)hi;
    return __builtin_shufflevector(a, b, 0, 1, 2, 3, 4, 5, 6, 7, 8, 9, 10, 11, 12, 13, 14, 15);
}

template <int K, bool STORE_HALF>
__global__ __launch_bounds__(256)
void gemm_bn_prelu_kernel(const _Float16* __restrict__ X, const _Float16* __restrict__ W,
                          const float* __restrict__ scale, const float* __restrict__ shift,
                          const float* __restrict__ a_ptr, void* __restrict__ outp) {
    __shared__ alignas(16) _Float16 As[2][128][40];   // 128x32 tiles, +8 pad, ping-pong
    __shared__ alignas(16) _Float16 Bs[2][128][40];

    const int tid  = threadIdx.x;
    const int lane = tid & 31;
    const int wave = tid >> 5;
    const int wm   = wave >> 1;        // 0..3 : 32-row band
    const int wn   = wave & 1;         // 0..1 : 64-col band
    const int r    = lane & 15;
    const int hi16 = lane >> 4;        // lane group 0/1 (ISA A/B K-split)

    const int m0 = blockIdx.x * 128;
    const int n0 = blockIdx.y * 128;

    // staging assignment: thread owns chunks {tid, tid+256} of each tile
    // chunk ch: row = ch>>2 (0..127), part = ch&3 (8-half / 16B units)
    const int ar = tid >> 2, ap = (tid & 3) * 8;
    const _Float16* aptr0 = X + (size_t)(m0 + ar)      * K + ap;
    const _Float16* aptr1 = X + (size_t)(m0 + 64 + ar) * K + ap;
    const _Float16* bptr0 = W + (size_t)(n0 + ar)      * K + ap;
    const _Float16* bptr1 = W + (size_t)(n0 + 64 + ar) * K + ap;

    v8f acc[2][4] = {};

    // prologue: stage k-step 0 into buffer 0
    *(uint4*)&As[0][ar][ap]      = *(const uint4*)aptr0;
    *(uint4*)&As[0][64 + ar][ap] = *(const uint4*)aptr1;
    *(uint4*)&Bs[0][ar][ap]      = *(const uint4*)bptr0;
    *(uint4*)&Bs[0][64 + ar][ap] = *(const uint4*)bptr1;
    __syncthreads();

    constexpr int NSTEP = K / 32;
    #pragma unroll 2
    for (int step = 0; step < NSTEP; ++step) {
        const int cur = step & 1, nxt = cur ^ 1;
        const bool has_next = (step + 1) < NSTEP;

        // issue global loads for the next tile early (overlap with WMMA)
        uint4 ra0, ra1, rb0, rb1;
        if (has_next) {
            const int ko = (step + 1) * 32;
            ra0 = *(const uint4*)(aptr0 + ko);
            ra1 = *(const uint4*)(aptr1 + ko);
            rb0 = *(const uint4*)(bptr0 + ko);
            rb1 = *(const uint4*)(bptr1 + ko);
        }

        // fragment loads from current buffer + 8 WMMAs
        v16h af[2], bf[4];
        #pragma unroll
        for (int fm = 0; fm < 2; fm++) {
            const _Float16* rowp = &As[cur][wm * 32 + fm * 16 + r][0];
            // ISA 16-bit A 16x32: lanes 0-15 hold K 0-7 & 16-23, lanes 16-31 K 8-15 & 24-31
            af[fm] = make_frag(rowp + hi16 * 8, rowp + hi16 * 8 + 16);
        }
        #pragma unroll
        for (int fn = 0; fn < 4; fn++) {
            const _Float16* colp = &Bs[cur][wn * 64 + fn * 16 + r][0];
            // ISA 16-bit B 32x16: lanes 0-15 hold K 0-7 & 16-23, lanes 16-31 K 8-15 & 24-31
            bf[fn] = make_frag(colp + hi16 * 8, colp + hi16 * 8 + 16);
        }
        #pragma unroll
        for (int fm = 0; fm < 2; fm++)
            #pragma unroll
            for (int fn = 0; fn < 4; fn++)
                acc[fm][fn] = __builtin_amdgcn_wmma_f32_16x16x32_f16(
                    false, af[fm], false, bf[fn], (short)0, acc[fm][fn], false, false);

        // stash the prefetched tile into the other buffer; reads of `nxt`
        // finished before the barrier that ended step-1, so this is safe.
        if (has_next) {
            *(uint4*)&As[nxt][ar][ap]      = ra0;
            *(uint4*)&As[nxt][64 + ar][ap] = ra1;
            *(uint4*)&Bs[nxt][ar][ap]      = rb0;
            *(uint4*)&Bs[nxt][64 + ar][ap] = rb1;
        }
        __syncthreads();
    }

    const float alpha = *a_ptr;
    #pragma unroll
    for (int fm = 0; fm < 2; fm++) {
        #pragma unroll
        for (int fn = 0; fn < 4; fn++) {
            const int nc = n0 + wn * 64 + fn * 16 + r;      // output channel
            const float sc = scale[nc], sh = shift[nc];
            const int mbase = m0 + wm * 32 + fm * 16 + hi16 * 8;
            #pragma unroll
            for (int t = 0; t < 8; t++) {
                float v = acc[fm][fn][t] * sc + sh;
                v = v >= 0.0f ? v : alpha * v;              // PReLU
                size_t o = (size_t)(mbase + t) * 256 + nc;
                if constexpr (STORE_HALF) ((_Float16*)outp)[o] = (_Float16)v;
                else                      ((float*)outp)[o]   = v;
            }
        }
    }
}

// ---------------------------------------------------------------------------
extern "C" void kernel_launch(void* const* d_in, const int* in_sizes, int n_in,
                              void* d_out, int out_size, void* d_ws, size_t ws_size,
                              hipStream_t stream) {
    const float* xyz1 = (const float*)d_in[0];
    const float* xyz2 = (const float*)d_in[1];
    const float* f1   = (const float*)d_in[2];
    const float* f2   = (const float*)d_in[3];
    const float* w1   = (const float*)d_in[4];
    const float* b1   = (const float*)d_in[5];
    const float* g1   = (const float*)d_in[6];
    const float* be1  = (const float*)d_in[7];
    const float* rm1  = (const float*)d_in[8];
    const float* rv1  = (const float*)d_in[9];
    const float* a1   = (const float*)d_in[10];
    const float* w2   = (const float*)d_in[11];
    const float* b2   = (const float*)d_in[12];
    const float* g2   = (const float*)d_in[13];
    const float* be2  = (const float*)d_in[14];
    const float* rm2  = (const float*)d_in[15];
    const float* rv2  = (const float*)d_in[16];
    const float* a2   = (const float*)d_in[17];

    char* ws = (char*)d_ws;
    size_t off = 0;
    _Float16* X1  = (_Float16*)(ws + off); off += (size_t)MROWS * 512 * sizeof(_Float16);
    _Float16* X2  = (_Float16*)(ws + off); off += (size_t)MROWS * 256 * sizeof(_Float16);
    _Float16* W1h = (_Float16*)(ws + off); off += (size_t)256 * 512 * sizeof(_Float16);
    _Float16* W2h = (_Float16*)(ws + off); off += (size_t)256 * 256 * sizeof(_Float16);
    float* scale1 = (float*)(ws + off); off += 256 * sizeof(float);
    float* shift1 = (float*)(ws + off); off += 256 * sizeof(float);
    float* scale2 = (float*)(ws + off); off += 256 * sizeof(float);
    float* shift2 = (float*)(ws + off); off += 256 * sizeof(float);

    bn_consts_kernel<<<1, 256, 0, stream>>>(b1, g1, be1, rm1, rv1, scale1, shift1);
    bn_consts_kernel<<<1, 256, 0, stream>>>(b2, g2, be2, rm2, rv2, scale2, shift2);
    f32_to_f16_kernel<<<(256 * 512 + 255) / 256, 256, 0, stream>>>(w1, W1h, 256 * 512);
    f32_to_f16_kernel<<<(256 * 256 + 255) / 256, 256, 0, stream>>>(w2, W2h, 256 * 256);

    interp_concat_kernel<<<MROWS / 8, 256, 0, stream>>>(xyz1, xyz2, f1, f2, X1);

    // N=256 covered by 2 blocks of 128 columns
    gemm_bn_prelu_kernel<512, true ><<<dim3(MROWS / 128, 2), 256, 0, stream>>>(
        X1, W1h, scale1, shift1, a1, (void*)X2);
    gemm_bn_prelu_kernel<256, false><<<dim3(MROWS / 128, 2), 256, 0, stream>>>(
        X2, W2h, scale2, shift2, a2, d_out);
}
